// CausalSelfAttention_46849503265302
// MI455X (gfx1250) — matrix operand order repaired and checked by reading, not compiled, not run on hardware
//
#include <hip/hip_runtime.h>
#include <hip/hip_bf16.h>
#include <math.h>

// Problem constants (reference: B=2, T=2048, C=1024, H=16, HD=64)
#define Bc  2
#define Tc  2048
#define Cc  1024
#define Hc  16
#define HDc 64
#define N3C (3 * Cc)
#define MTOT (Bc * Tc)          // 4096 rows

typedef __attribute__((ext_vector_type(16))) __bf16 v16bf;
typedef __attribute__((ext_vector_type(8)))  __bf16 v8bf;
typedef __attribute__((ext_vector_type(8)))  float  v8f;

#define WMMA_BF16(a, b, c) \
  __builtin_amdgcn_wmma_f32_16x16x32_bf16(false, (a), false, (b), (short)0, (c), false, false)

static __device__ __forceinline__ v16bf cat8(v8bf lo, v8bf hi) {
  return __builtin_shufflevector(lo, hi, 0,1,2,3,4,5,6,7,8,9,10,11,12,13,14,15);
}

// A fragment, 16x32 bf16, source row-major [M][K].
// Contiguous k order used here (not the ISA table order): lane holds row M=lane&15; element i -> K = kbase + 16*hf + i
static __device__ __forceinline__ v16bf load_a_frag(const __bf16* row, int kbase, int hf) {
  v8bf lo = *(const v8bf*)(row + kbase + 16 * hf);
  v8bf hi = *(const v8bf*)(row + kbase + 16 * hf + 8);
  return cat8(lo, hi);
}

// B fragment, 32x16 bf16, source K-contiguous per column (ptr = &Bsrc[col][0]).
// Contiguous k order used here (not the ISA table order): lane holds col N=lane&15; element i -> K = kbase + 16*hf + i
static __device__ __forceinline__ v16bf load_b_frag(const __bf16* col, int kbase, int hf) {
  v8bf lo = *(const v8bf*)(col + kbase + 16 * hf);
  v8bf hi = *(const v8bf*)(col + kbase + 16 * hf + 8);
  return cat8(lo, hi);
}

// CDNA5 async copy: 16B global -> LDS, no VGPR roundtrip, tracked by ASYNCcnt.
static __device__ __forceinline__ void async_copy16(uint32_t lds_byte_off, const void* gptr) {
  asm volatile("global_load_async_to_lds_b128 %0, %1, off"
               :: "v"(lds_byte_off), "v"((unsigned long long)(uintptr_t)gptr)
               : "memory");
}
static __device__ __forceinline__ void wait_async0() {
#if __has_builtin(__builtin_amdgcn_s_wait_asynccnt)
  __builtin_amdgcn_s_wait_asynccnt(0);
#else
  asm volatile("s_wait_asynccnt 0x0" ::: "memory");
#endif
}

// Row reductions across a 16-lane half (C/D rows live in one half on CDNA5 wave32).
static __device__ __forceinline__ float redmax16(float v) {
#pragma unroll
  for (int m = 1; m < 16; m <<= 1) v = fmaxf(v, __shfl_xor(v, m, 32));
  return v;
}
static __device__ __forceinline__ float redsum16(float v) {
#pragma unroll
  for (int m = 1; m < 16; m <<= 1) v += __shfl_xor(v, m, 32);
  return v;
}

// ---------------- conversion kernels ----------------
__global__ void cvt_f32_bf16(const float* __restrict__ src, __bf16* __restrict__ dst, int n) {
  int i = blockIdx.x * 256 + threadIdx.x;
  if (i < n) dst[i] = (__bf16)src[i];
}
// dst[n*K + k] = src[k*N + n]  (K-major / transposed weights for B fragments)
__global__ void cvt_transpose_bf16(const float* __restrict__ src, __bf16* __restrict__ dst,
                                   int K, int N) {
  int i = blockIdx.x * 256 + threadIdx.x;
  if (i < K * N) {
    int n = i / K, k = i % K;
    dst[i] = (__bf16)src[(size_t)k * N + n];
  }
}

// LDS B tile: 64 columns x 32 K, padded column stride 48 bf16 (96B, 16B-aligned, bank-spread)
#define BSTRIDE 48
#define BTILE_ELTS (64 * BSTRIDE)

// ---------------- QKV GEMM ----------------
// Block = 8 waves stacked along M: block tile 128(M) x 64(N), shared B tile staged
// into LDS via async-to-LDS (double buffered). Wave tile 16x64: 1 A frag -> 4 WMMAs.
__global__ __launch_bounds__(256) void qkv_gemm(const __bf16* __restrict__ xb,
                                                const __bf16* __restrict__ wT,
                                                const float* __restrict__ bias,
                                                __bf16* __restrict__ qb,
                                                __bf16* __restrict__ kb,
                                                __bf16* __restrict__ vtb) {
  __shared__ __bf16 sB[2][BTILE_ELTS];             // 2 x 6KB
  const int lane = threadIdx.x;
  const int wave = threadIdx.y;
  const int hf = lane >> 4;
  const int ln = lane & 15;
  const int t256 = wave * 32 + lane;
  const int n0 = blockIdx.x * 64;
  const int m0 = blockIdx.y * 128 + wave * 16;

  const __bf16* arow = xb + (size_t)(m0 + ln) * Cc;
  // this thread's 16B chunk of the staged B tile
  const int cn = t256 >> 2;                        // 0..63 (column)
  const int ck = (t256 & 3) * 8;                   // 0,8,16,24 (k within tile)
  const __bf16* gsrc = wT + (size_t)(n0 + cn) * Cc + ck;

  async_copy16((uint32_t)(uintptr_t)&sB[0][cn * BSTRIDE + ck], gsrc);
  wait_async0();
  __syncthreads();

  v8f acc0 = {}, acc1 = {}, acc2 = {}, acc3 = {};
  for (int kk = 0; kk < Cc / 32; kk++) {
    const int buf = kk & 1;
    if (kk + 1 < Cc / 32)
      async_copy16((uint32_t)(uintptr_t)&sB[buf ^ 1][cn * BSTRIDE + ck],
                   gsrc + (kk + 1) * 32);
    if (kk + 16 < Cc / 32) __builtin_prefetch(arow + (kk + 16) * 32, 0, 3);

    v16bf a = load_a_frag(arow, kk * 32, hf);
    const __bf16* sb = &sB[buf][16 * hf];
    v16bf b0 = cat8(*(const v8bf*)(sb + (ln +  0) * BSTRIDE),
                    *(const v8bf*)(sb + (ln +  0) * BSTRIDE + 8));
    v16bf b1 = cat8(*(const v8bf*)(sb + (ln + 16) * BSTRIDE),
                    *(const v8bf*)(sb + (ln + 16) * BSTRIDE + 8));
    v16bf b2 = cat8(*(const v8bf*)(sb + (ln + 32) * BSTRIDE),
                    *(const v8bf*)(sb + (ln + 32) * BSTRIDE + 8));
    v16bf b3 = cat8(*(const v8bf*)(sb + (ln + 48) * BSTRIDE),
                    *(const v8bf*)(sb + (ln + 48) * BSTRIDE + 8));
    acc0 = WMMA_BF16(a, b0, acc0);
    acc1 = WMMA_BF16(a, b1, acc1);
    acc2 = WMMA_BF16(a, b2, acc2);
    acc3 = WMMA_BF16(a, b3, acc3);

    wait_async0();       // next buffer resident before anyone proceeds
    __syncthreads();     // and everyone done reading current buffer
  }

#pragma unroll
  for (int dt = 0; dt < 4; dt++) {
    const v8f acc = dt == 0 ? acc0 : dt == 1 ? acc1 : dt == 2 ? acc2 : acc3;
    const int n = n0 + dt * 16 + ln;
    const float bn = bias[n];
    const int seg = n >> 10;                       // 0=Q 1=K 2=V (uniform per block)
    const int nn = n & 1023;
    const int h = nn >> 6, d = nn & 63;
#pragma unroll
    for (int r = 0; r < 8; r++) {
      const int m = m0 + r + 8 * hf;               // C/D layout: row = r + 8*half
      const int bb = m >> 11, t = m & (Tc - 1);
      const int bh = bb * Hc + h;
      const __bf16 val = (__bf16)(acc[r] + bn);
      if (seg == 0)      qb[((size_t)bh * Tc + t) * HDc + d] = val;
      else if (seg == 1) kb[((size_t)bh * Tc + t) * HDc + d] = val;
      else               vtb[((size_t)bh * HDc + d) * Tc + t] = val; // V stored [bh][d][t]
    }
  }
}

// ---------------- flash attention: per-wave 16-query tile, key tiles of 32 ----------------
__global__ __launch_bounds__(256) void flash_attn(const __bf16* __restrict__ qb,
                                                  const __bf16* __restrict__ kb,
                                                  const __bf16* __restrict__ vtb,
                                                  __bf16* __restrict__ yb) {
  __shared__ float sP[8 * 16 * 32];                // 16KB: per-wave 16x32 P tile
  const int lane = threadIdx.x;
  const int wave = threadIdx.y;
  const int hf = lane >> 4;
  const int ln = lane & 15;
  const int q0 = (blockIdx.x * 8 + wave) * 16;
  const int bh = blockIdx.y;
  const int b = bh / Hc, h = bh % Hc;
  float* sp = sP + wave * 512;

  const __bf16* qrow = qb + ((size_t)bh * Tc + q0 + ln) * HDc;
  const v16bf qa0 = load_a_frag(qrow, 0, hf);
  const v16bf qa1 = load_a_frag(qrow, 32, hf);

  v8f o0 = {}, o1 = {}, o2 = {}, o3 = {};
  float mrun[8], lrun[8];
#pragma unroll
  for (int r = 0; r < 8; r++) { mrun[r] = -1e30f; lrun[r] = 0.0f; }
  const float scale = 0.125f;                      // 1/sqrt(64)

  for (int kt = 0; kt < q0 + 16; kt += 32) {
    const __bf16* kr0 = kb + ((size_t)bh * Tc + kt + ln) * HDc;
    const __bf16* kr1 = kr0 + 16 * HDc;
    v8f s0 = {}, s1 = {};
    s0 = WMMA_BF16(qa0, load_b_frag(kr0, 0, hf), s0);
    s0 = WMMA_BF16(qa1, load_b_frag(kr0, 32, hf), s0);
    s1 = WMMA_BF16(qa0, load_b_frag(kr1, 0, hf), s1);
    s1 = WMMA_BF16(qa1, load_b_frag(kr1, 32, hf), s1);

    float p0[8], p1[8], corr[8];
#pragma unroll
    for (int r = 0; r < 8; r++) {
      const int qg = q0 + r + 8 * hf;
      float a = s0[r] * scale; if (kt + ln > qg)      a = -1e30f;
      float c = s1[r] * scale; if (kt + 16 + ln > qg) c = -1e30f;
      float mt = redmax16(fmaxf(a, c));
      const float mnew = fmaxf(mrun[r], mt);
      const float cr = __expf(mrun[r] - mnew);
      const float e0 = __expf(a - mnew);
      const float e1 = __expf(c - mnew);
      lrun[r] = lrun[r] * cr + redsum16(e0 + e1);
      mrun[r] = mnew;
      corr[r] = cr;
      p0[r] = e0; p1[r] = e1;
    }
#pragma unroll
    for (int r = 0; r < 8; r++) {
      o0[r] *= corr[r]; o1[r] *= corr[r]; o2[r] *= corr[r]; o3[r] *= corr[r];
    }

    // transpose P (C/D layout -> A fragment) through LDS; same-wave DS ops are in-order
#pragma unroll
    for (int r = 0; r < 8; r++) {
      sp[(r + 8 * hf) * 32 + ln] = p0[r];
      sp[(r + 8 * hf) * 32 + 16 + ln] = p1[r];
    }
    const float* prow = sp + ln * 32;
    v16bf pf;
#pragma unroll
    for (int i = 0; i < 8; i++) pf[i] = (__bf16)prow[16 * hf + i];
#pragma unroll
    for (int i = 0; i < 8; i++) pf[8 + i] = (__bf16)prow[16 * hf + 8 + i];

    const __bf16* vcol = vtb + ((size_t)bh * HDc + ln) * Tc + kt;
    o0 = WMMA_BF16(pf, load_b_frag(vcol + (size_t)0 * 16 * Tc, 0, hf), o0);
    o1 = WMMA_BF16(pf, load_b_frag(vcol + (size_t)1 * 16 * Tc, 0, hf), o1);
    o2 = WMMA_BF16(pf, load_b_frag(vcol + (size_t)2 * 16 * Tc, 0, hf), o2);
    o3 = WMMA_BF16(pf, load_b_frag(vcol + (size_t)3 * 16 * Tc, 0, hf), o3);
  }

#pragma unroll
  for (int r = 0; r < 8; r++) {
    const float inv = 1.0f / lrun[r];
    const int t = q0 + r + 8 * hf;
    const size_t base = ((size_t)b * Tc + t) * Cc + h * HDc + ln;
    yb[base + 0]  = (__bf16)(o0[r] * inv);
    yb[base + 16] = (__bf16)(o1[r] * inv);
    yb[base + 32] = (__bf16)(o2[r] * inv);
    yb[base + 48] = (__bf16)(o3[r] * inv);
  }
}

// ---------------- projection GEMM: same blocking, fp32 epilogue into d_out ----------------
__global__ __launch_bounds__(256) void proj_gemm(const __bf16* __restrict__ yb,
                                                 const __bf16* __restrict__ wT,
                                                 const float* __restrict__ bias,
                                                 float* __restrict__ out) {
  __shared__ __bf16 sB[2][BTILE_ELTS];
  const int lane = threadIdx.x;
  const int wave = threadIdx.y;
  const int hf = lane >> 4;
  const int ln = lane & 15;
  const int t256 = wave * 32 + lane;
  const int n0 = blockIdx.x * 64;
  const int m0 = blockIdx.y * 128 + wave * 16;

  const __bf16* arow = yb + (size_t)(m0 + ln) * Cc;
  const int cn = t256 >> 2;
  const int ck = (t256 & 3) * 8;
  const __bf16* gsrc = wT + (size_t)(n0 + cn) * Cc + ck;

  async_copy16((uint32_t)(uintptr_t)&sB[0][cn * BSTRIDE + ck], gsrc);
  wait_async0();
  __syncthreads();

  v8f acc0 = {}, acc1 = {}, acc2 = {}, acc3 = {};
  for (int kk = 0; kk < Cc / 32; kk++) {
    const int buf = kk & 1;
    if (kk + 1 < Cc / 32)
      async_copy16((uint32_t)(uintptr_t)&sB[buf ^ 1][cn * BSTRIDE + ck],
                   gsrc + (kk + 1) * 32);
    if (kk + 16 < Cc / 32) __builtin_prefetch(arow + (kk + 16) * 32, 0, 3);

    v16bf a = load_a_frag(arow, kk * 32, hf);
    const __bf16* sb = &sB[buf][16 * hf];
    v16bf b0 = cat8(*(const v8bf*)(sb + (ln +  0) * BSTRIDE),
                    *(const v8bf*)(sb + (ln +  0) * BSTRIDE + 8));
    v16bf b1 = cat8(*(const v8bf*)(sb + (ln + 16) * BSTRIDE),
                    *(const v8bf*)(sb + (ln + 16) * BSTRIDE + 8));
    v16bf b2 = cat8(*(const v8bf*)(sb + (ln + 32) * BSTRIDE),
                    *(const v8bf*)(sb + (ln + 32) * BSTRIDE + 8));
    v16bf b3 = cat8(*(const v8bf*)(sb + (ln + 48) * BSTRIDE),
                    *(const v8bf*)(sb + (ln + 48) * BSTRIDE + 8));
    acc0 = WMMA_BF16(a, b0, acc0);
    acc1 = WMMA_BF16(a, b1, acc1);
    acc2 = WMMA_BF16(a, b2, acc2);
    acc3 = WMMA_BF16(a, b3, acc3);

    wait_async0();
    __syncthreads();
  }

#pragma unroll
  for (int dt = 0; dt < 4; dt++) {
    const v8f acc = dt == 0 ? acc0 : dt == 1 ? acc1 : dt == 2 ? acc2 : acc3;
    const int n = n0 + dt * 16 + ln;
    const float bn = bias[n];
#pragma unroll
    for (int r = 0; r < 8; r++) {
      const int m = m0 + r + 8 * hf;
      out[(size_t)m * Cc + n] = acc[r] + bn;
    }
  }
}

extern "C" void kernel_launch(void* const* d_in, const int* in_sizes, int n_in,
                              void* d_out, int out_size, void* d_ws, size_t ws_size,
                              hipStream_t stream) {
  (void)in_sizes; (void)n_in; (void)out_size; (void)ws_size;
  const float* x      = (const float*)d_in[0];   // [2,2048,1024]
  const float* w_attn = (const float*)d_in[1];   // [1024,3072]
  const float* b_attn = (const float*)d_in[2];   // [3072]
  const float* w_proj = (const float*)d_in[3];   // [1024,1024]
  const float* b_proj = (const float*)d_in[4];   // [1024]
  float* out = (float*)d_out;                    // [2,2048,1024] fp32

  char* ws = (char*)d_ws;
  __bf16* xb   = (__bf16*)(ws + 0);                   //  8 MB  x bf16 [4096,1024]
  __bf16* watT = (__bf16*)(ws + 8388608);             //  6 MB  w_attn^T bf16 [3072,1024]
  __bf16* wpjT = (__bf16*)(ws + 14680064);            //  2 MB  w_proj^T bf16 [1024,1024]
  __bf16* qb   = (__bf16*)(ws + 16777216);            //  8 MB  Q  [32,2048,64]
  __bf16* kb   = (__bf16*)(ws + 25165824);            //  8 MB  K  [32,2048,64]
  __bf16* vtb  = (__bf16*)(ws + 33554432);            //  8 MB  V^T [32,64,2048]
  __bf16* ybb  = (__bf16*)(ws + 41943040);            //  8 MB  y  [4096,1024]

  const int nx  = MTOT * Cc;         // 4194304
  const int nwa = Cc * N3C;          // 3145728
  const int nwp = Cc * Cc;           // 1048576

  cvt_f32_bf16<<<(nx + 255) / 256, 256, 0, stream>>>(x, xb, nx);
  cvt_transpose_bf16<<<(nwa + 255) / 256, 256, 0, stream>>>(w_attn, watT, Cc, N3C);
  cvt_transpose_bf16<<<(nwp + 255) / 256, 256, 0, stream>>>(w_proj, wpjT, Cc, Cc);

  // block tile 128x64, grid (N/64, M/128)
  qkv_gemm<<<dim3(N3C / 64, MTOT / 128), dim3(32, 8), 0, stream>>>(xb, watT, b_attn, qb, kb, vtb);

  // grid: x = 128 q-tiles / 8 waves, y = B*H = 32
  flash_attn<<<dim3(16, 32), dim3(32, 8), 0, stream>>>(qb, kb, vtb, ybb);

  proj_gemm<<<dim3(Cc / 64, MTOT / 128), dim3(32, 8), 0, stream>>>(ybb, wpjT, b_proj, out);
}
